// HistogramLoss_67748814127604
// MI455X (gfx1250) — matrix operand order repaired and checked by reading, not compiled, not run on hardware
//
#include <hip/hip_runtime.h>
#include <hip/hip_bf16.h>
#include <math.h>

// ---------------------------------------------------------------------------
// HistogramLoss for gfx1250 (MI455X).
//
// feature: [1,64,128,128] f32  (nearest-upsampled 4x -> 512x512 logically)
// label:   [1,1,512,512]  int
// out:     scalar f32
//
// Pipeline:
//   K0 zero scratch accumulators
//   K1 weights: W[cp][c] = #fine labels==c in 4x4 block (f16, classes padded to 32)
//   K2 WMMA:   S1[c][d] = sum_p W*feat,  S2[c][d] = sum_p W*feat^2
//              (M=64 dims, N=32 classes, K=16384 coarse pixels, f16 in / f32 acc)
//   K3 stats:  n[c] (reduce W), miu, inv_sigma_s = 5/std
//   K4 KDE:    per-COARSE-pixel weighted exp accumulation (exact: all fine
//              pixels of a 4x4 block share feat + stats -> weight w=count).
//              stats table staged to LDS via TENSOR_LOAD_TO_LDS (TDM).
//   K5 loss:   normalize hist, smooth-L1 vs constant target, active average
// ---------------------------------------------------------------------------

typedef __attribute__((ext_vector_type(16))) _Float16 v16h;
typedef __attribute__((ext_vector_type(8)))  float    v8f;
typedef __attribute__((ext_vector_type(4)))  unsigned int tdm_v4u;
typedef __attribute__((ext_vector_type(8)))  int          tdm_v8i;
typedef __attribute__((ext_vector_type(4)))  int          tdm_v4i;

#define DD        64          // feature dims
#define NC        19          // classes
#define NCP       32          // padded classes
#define CP        16384       // coarse pixels (128*128)
#define HIST_N    (NC*DD*7)   // 8512

// float-offset workspace layout
#define OFF_S1    0           // 32*64            = 2048
#define OFF_S2    2048        // 32*64            = 2048
#define OFF_N     4096        // 32
#define OFF_STATS 4128        // 32*64*2          = 4096 (miu, inv_sigma_s)
#define OFF_HIST  8224        // 19*64*7          = 8512
#define ZERO_CNT  16736
#define OFF_W     16768       // f16 W[16384][32] = 1 MB

__global__ void hl_zero_kernel(float* ws, int n) {
  int i = blockIdx.x * blockDim.x + threadIdx.x;
  if (i < n) ws[i] = 0.0f;
}

// One thread per coarse pixel: count labels in its 4x4 fine block.
__global__ void hl_weights_kernel(const int* __restrict__ lab,
                                  _Float16* __restrict__ W) {
  int cp = blockIdx.x * blockDim.x + threadIdx.x;
  if (cp >= CP) return;
  int cy = cp >> 7, cx = cp & 127;
  int l[16];
#pragma unroll
  for (int i = 0; i < 16; ++i) {
    int fy = cy * 4 + (i >> 2);
    int fx = cx * 4 + (i & 3);
    l[i] = lab[fy * 512 + fx];
  }
#pragma unroll 4
  for (int c = 0; c < NCP; ++c) {
    int cnt = 0;
#pragma unroll
    for (int i = 0; i < 16; ++i) cnt += (l[i] == c) ? 1 : 0;
    W[(size_t)cp * NCP + c] = (_Float16)(float)cnt;
  }
}

// WMMA GEMM: D[m][n] = sum_K A[m][K]*B[K][n].
// grid = (4 m-tiles, 2 n-tiles, 64 k-blocks), block = 32 (one wave, EXEC all-1s).
__global__ void hl_wmma_sums_kernel(const float* __restrict__ feat,
                                    const _Float16* __restrict__ W,
                                    float* __restrict__ S1,
                                    float* __restrict__ S2) {
  const int m0   = blockIdx.x * 16;
  const int n0   = blockIdx.y * 16;
  const int lane = threadIdx.x;           // 0..31
  const int MN   = lane & 15;             // A row / B col for this lane
  const int hi   = (lane >> 4) & 1;       // upper lane half
  const int kh8  = hi ? 8 : 0;            // A-layout K offset
  const int kh16 = hi ? 16 : 0;           // unused; B uses A's kh8 K order (Kb = Ka)

  v8f acc1 = {};
  v8f acc2 = {};

  for (int kc = 0; kc < 8; ++kc) {
    const int kbase = blockIdx.z * 256 + kc * 32;
    v16h a1, a2, b;
#pragma unroll
    for (int e = 0; e < 16; ++e) {
      // 16-bit A 16x32 layout: VGPR v(0..3): K = kh8 + 2v,2v+1 ; v(4..7): +16
      const int vg  = e >> 1;
      const int Ka  = ((vg >> 2) << 4) + kh8 + ((vg & 3) << 1) + (e & 1);
      // 16-bit B 32x16 layout: same K order as A
      const int Kb  = Ka;
      const float fx = feat[(size_t)(m0 + MN) * CP + (kbase + Ka)];
      a1[e] = (_Float16)fx;
      a2[e] = (_Float16)(fx * fx);
      b[e]  = W[(size_t)(kbase + Kb) * NCP + (n0 + MN)];
    }
    acc1 = __builtin_amdgcn_wmma_f32_16x16x32_f16(false, a1, false, b,
                                                  (short)0, acc1, false, false);
    acc2 = __builtin_amdgcn_wmma_f32_16x16x32_f16(false, a2, false, b,
                                                  (short)0, acc2, false, false);
  }

  // C/D layout: VGPR r holds row M = r + (hi?8:0), col N = lane&15.
#pragma unroll
  for (int r = 0; r < 8; ++r) {
    const int m = m0 + r + (hi ? 8 : 0);
    const int c = n0 + MN;
    atomicAdd(&S1[c * DD + m], acc1[r]);
    atomicAdd(&S2[c * DD + m], acc2[r]);
  }
}

// Per class: n[c] = sum_p W[p][c]; miu/var -> (miu, 5/std) table.
__global__ void hl_stats_kernel(const _Float16* __restrict__ W,
                                const float* __restrict__ S1,
                                const float* __restrict__ S2,
                                float* __restrict__ nArr,
                                float* __restrict__ stats) {
  const int c = blockIdx.x;               // 0..18
  __shared__ float red[256];
  float s = 0.0f;
  for (int p = threadIdx.x; p < CP; p += 256)
    s += (float)W[(size_t)p * NCP + c];
  red[threadIdx.x] = s;
  __syncthreads();
  for (int st = 128; st > 0; st >>= 1) {
    if (threadIdx.x < st) red[threadIdx.x] += red[threadIdx.x + st];
    __syncthreads();
  }
  const float n = red[0];
  if (threadIdx.x == 0) nArr[c] = n;
  const float n_safe = fmaxf(n, 1.0f);

  for (int d = threadIdx.x; d < DD; d += 256) {
    const float s1  = S1[c * DD + d];
    const float s2  = S2[c * DD + d];
    const float miu = s1 / n_safe;
    // sum mask*(x-miu)^2 = S2 - 2*miu*S1 + miu^2*n
    const float var = (s2 - 2.0f * miu * s1 + miu * miu * n) / n_safe + 1e-10f;
    stats[(c * DD + d) * 2 + 0] = miu;
    stats[(c * DD + d) * 2 + 1] = 5.0f * rsqrtf(var);   // 1/sigma_s
  }
}

// KDE accumulation over COARSE pixels. All 16 fine pixels of a 4x4 block share
// the same feature column, so the per-(c,d,k) Gaussian term is identical for
// every fine pixel of that class in the block: contribution = W[cp][c] * exp.
// z = (miu-x)/sigma_s + 5k; the 1/sqrt(2*pi*var_s) prefactor cancels in the
// bin normalization. Stats table staged into LDS with the Tensor Data Mover.
__global__ void hl_kde_kernel(const float* __restrict__ feat,
                              const _Float16* __restrict__ W,
                              const float* __restrict__ stats,
                              float* __restrict__ hist) {
  __shared__ float lhist[HIST_N];          // 34 KB
  __shared__ float lstats[NC * DD * 2];    // 9.5 KB
  for (int i = threadIdx.x; i < HIST_N; i += blockDim.x) lhist[i] = 0.0f;

#if defined(__gfx1250__) && __has_builtin(__builtin_amdgcn_tensor_load_to_lds)
  if (threadIdx.x == 0) {
    // D# per CDNA5 ISA: 1-row tile of 2432 f32 elements.
    const unsigned long long ga = (unsigned long long)(uintptr_t)stats;
    const unsigned lds = (unsigned)(uintptr_t)lstats;   // LDS byte offset
    const unsigned td0 = (unsigned)(NC * DD * 2);       // 2432 elements (4B each)
    tdm_v4u g0 = { 1u,                                  // count=1 user descriptor
                   lds,                                  // lds_addr [63:32]
                   (unsigned)(ga & 0xffffffffu),         // global_addr lo
                   (unsigned)((ga >> 32) & 0x1ffffffu)   // global_addr hi [120:96]
                     | (2u << 30) };                     // type=2 (image)
    tdm_v8i g1 = { (int)(2u << 16),                      // data_size=2 -> 4 bytes
                   (int)((td0 & 0xffffu) << 16),         // tensor_dim0[15:0] @63:48
                   (int)((td0 >> 16) | (1u << 16)),      // dim0 hi | tensor_dim1=1
                   (int)(td0 << 16),                     // tile_dim0 @127:112
                   (int)1,                               // tile_dim1=1
                   (int)td0,                             // tensor_dim0_stride lo
                   0, 0 };
    tdm_v4i g2 = { 0, 0, 0, 0 };
    tdm_v4i g3 = { 0, 0, 0, 0 };
#if __clang_major__ >= 23
    tdm_v8i g4 = { 0, 0, 0, 0, 0, 0, 0, 0 };
    __builtin_amdgcn_tensor_load_to_lds(g0, g1, g2, g3, g4, 0);
#else
    __builtin_amdgcn_tensor_load_to_lds(g0, g1, g2, g3, 0);
#endif
  }
  __builtin_amdgcn_s_wait_tensorcnt(0);
#else
  for (int i = threadIdx.x; i < NC * DD * 2; i += blockDim.x) lstats[i] = stats[i];
#endif
  __syncthreads();

  const int cp = blockIdx.x * blockDim.x + threadIdx.x;   // one coarse pixel
  if (cp < CP) {
    for (int c = 0; c < NC; ++c) {
      const float w = (float)W[(size_t)cp * NCP + c];
      if (w == 0.0f) continue;
      const float* st = &lstats[c * DD * 2];
      float* hc = &lhist[c * DD * 7];
      for (int d = 0; d < DD; ++d) {
        const float x = feat[(size_t)d * CP + cp];
        const float u = (st[d * 2] - x) * st[d * 2 + 1];
#pragma unroll
        for (int k = 0; k < 7; ++k) {
          const float z  = u + 5.0f * (float)(k - 3);
          const float zz = z * z;
          if (zz < 50.0f)                 // e^-25 ~ 1e-11: negligible after norm
            atomicAdd(&hc[d * 7 + k], w * __expf(-0.5f * zz));
        }
      }
    }
  }
  __syncthreads();
  for (int i = threadIdx.x; i < HIST_N; i += blockDim.x)
    if (lhist[i] != 0.0f) atomicAdd(&hist[i], lhist[i]);
}

// Normalize + smooth-L1 vs constant target, active-weighted average.
__global__ void hl_loss_kernel(const float* __restrict__ hist,
                               const float* __restrict__ nArr,
                               float* __restrict__ out) {
  __shared__ float red[256];
  __shared__ float act[NC];
  float t[7];
  float tsum = 0.0f;
#pragma unroll
  for (int k = 0; k < 7; ++k) {
    const float kk = (float)(k - 3);
    t[k] = __expf(-0.5f * kk * kk);
    tsum += t[k];
  }
#pragma unroll
  for (int k = 0; k < 7; ++k) t[k] /= tsum;

  if (threadIdx.x < NC)
    act[threadIdx.x] = (nArr[threadIdx.x] >= 1000.0f) ? 1.0f : 0.0f;
  __syncthreads();

  float partial = 0.0f;
  for (int cd = threadIdx.x; cd < NC * DD; cd += blockDim.x) {
    const float a = act[cd / DD];
    if (a == 0.0f) continue;
    const float* h = &hist[cd * 7];
    float s = 0.0f;
#pragma unroll
    for (int k = 0; k < 7; ++k) s += h[k];
    const float invs = 1.0f / fmaxf(s, 1e-30f);
    float l = 0.0f;
#pragma unroll
    for (int k = 0; k < 7; ++k) {
      const float d = fabsf(h[k] * invs - t[k]);
      l += (d < 1.0f) ? 0.5f * d * d : (d - 0.5f);
    }
    partial += a * l * (1.0f / 448.0f);    // mean over D*7 elements per class
  }
  red[threadIdx.x] = partial;
  __syncthreads();
  for (int st = 128; st > 0; st >>= 1) {
    if (threadIdx.x < st) red[threadIdx.x] += red[threadIdx.x + st];
    __syncthreads();
  }
  if (threadIdx.x == 0) {
    float asum = 0.0f;
    for (int c = 0; c < NC; ++c) asum += act[c];
    out[0] = red[0] / fmaxf(asum, 1e-30f);  // LOSS_WEIGHT = 1
  }
}

extern "C" void kernel_launch(void* const* d_in, const int* in_sizes, int n_in,
                              void* d_out, int out_size, void* d_ws, size_t ws_size,
                              hipStream_t stream) {
  (void)in_sizes; (void)n_in; (void)out_size; (void)ws_size;
  const float* feat = (const float*)d_in[0];   // [1,64,128,128]
  const int*   lab  = (const int*)d_in[1];     // [1,1,512,512]

  float* ws     = (float*)d_ws;                // needs ~1.12 MB scratch
  float* S1     = ws + OFF_S1;
  float* S2     = ws + OFF_S2;
  float* nArr   = ws + OFF_N;
  float* stats  = ws + OFF_STATS;
  float* hist   = ws + OFF_HIST;
  _Float16* W   = (_Float16*)(ws + OFF_W);

  hl_zero_kernel<<<(ZERO_CNT + 255) / 256, 256, 0, stream>>>(ws, ZERO_CNT);
  hl_weights_kernel<<<CP / 256, 256, 0, stream>>>(lab, W);
  dim3 gw(4, 2, 64);
  hl_wmma_sums_kernel<<<gw, 32, 0, stream>>>(feat, W, S1, S2);
  hl_stats_kernel<<<NC, 256, 0, stream>>>(W, S1, S2, nArr, stats);
  hl_kde_kernel<<<CP / 256, 256, 0, stream>>>(feat, W, stats, hist);
  hl_loss_kernel<<<1, 256, 0, stream>>>(hist, nArr, (float*)d_out);
}
